// FourSequenceMHAttention_21148418965765
// MI455X (gfx1250) — matrix operand order repaired and checked by reading, not compiled, not run on hardware
//
#include <hip/hip_runtime.h>

typedef __attribute__((ext_vector_type(16))) _Float16 v16h;
typedef __attribute__((ext_vector_type(8)))  _Float16 v8h;
typedef __attribute__((ext_vector_type(8)))  float    v8f;
typedef __attribute__((ext_vector_type(4)))  unsigned int v4u;
typedef __attribute__((ext_vector_type(8)))  int v8i;
typedef __attribute__((ext_vector_type(4)))  int v4i;

// ---------------------------------------------------------------------------
// WMMA fragment helpers (CDNA5 wave32 layouts, CDNA5 ISA §7.12.2)
// A (16x32 f16): lane -> M = lane&15; elem i<8 -> K = (lane>>4)*8+i,
//                                     elem i>=8 -> K = 16+(lane>>4)*8+(i-8)
// B (32x16 f16): lane -> N = lane&15; elem i -> K same as A's elem i
// C/D (16x16 f32): elem r -> M = r + 8*(lane>>4); N = lane&15
// ---------------------------------------------------------------------------
__device__ __forceinline__ v16h load_a_frag(const _Float16* tile, int ld) {
  const int lane = threadIdx.x & 31;
  const int m = lane & 15;
  const int hh = lane >> 4;
  const _Float16* row = tile + (size_t)m * ld;
  v8h lo = *(const v8h*)(row + hh * 8);
  v8h hi = *(const v8h*)(row + 16 + hh * 8);
  v16h a;
#pragma unroll
  for (int i = 0; i < 8; ++i) { a[i] = lo[i]; a[8 + i] = hi[i]; }
  return a;
}

__device__ __forceinline__ v16h load_b_frag(const _Float16* tile, int ld) {
  const int lane = threadIdx.x & 31;
  const int n = lane & 15;
  const int hh = lane >> 4;
  v16h b;
#pragma unroll
  for (int i = 0; i < 8; ++i) {
    b[i] = tile[(size_t)(hh * 8 + i) * ld + n];
    b[8 + i] = tile[(size_t)(16 + hh * 8 + i) * ld + n];
  }
  return b;
}

__device__ __forceinline__ v8f wmma_f16(v16h a, v16h b, v8f c) {
  return __builtin_amdgcn_wmma_f32_16x16x32_f16(false, a, false, b, (short)0, c,
                                                false, false);
}

// ---------------------------------------------------------------------------
// Tensor Data Mover: DMA a 32-row x 128-f16 tile (row-major, contiguous rows)
// from global into LDS with a 272B destination pitch (pad 16B after each 256B
// row).  D# per CDNA5 ISA.  Issued per-wave, tracked
// with TENSORcnt.
// ---------------------------------------------------------------------------
__device__ __forceinline__ void tdm_load_tile_32x128_f16(unsigned lds_byte_addr,
                                                         const _Float16* gsrc) {
  const unsigned long long ga = (unsigned long long)(size_t)gsrc;
  v4u g0;
  g0[0] = 1u;                                    // count=1, user descriptor
  g0[1] = lds_byte_addr;                         // lds_addr
  g0[2] = (unsigned)ga;                          // global_addr[31:0]
  g0[3] = (unsigned)((ga >> 32) & 0x01ffffffu)   // global_addr[56:32]
          | (2u << 30);                          // type = 2 ("image")
  v8i g1;
  g1[0] = (int)((1u << 16)     // data_size = 2 bytes
              | (1u << 20)     // pad_enable
              | (5u << 22)     // pad_interval: 64 DWORDs (256B) between pads
              | (3u << 25));   // pad_amount:   4 DWORDs (16B) -> 272B pitch
  g1[1] = (int)(128u << 16);   // tensor_dim0 = 128 elements
  g1[2] = (int)(32u << 16);    // tensor_dim1 = 32 rows
  g1[3] = (int)(128u << 16);   // tile_dim0 = 128
  g1[4] = 32;                  // tile_dim1 = 32 (tile_dim2 = 0)
  g1[5] = 128;                 // tensor_dim0_stride = 128 elements
  g1[6] = 0;
  g1[7] = 0;
  v4i z4 = {};
#if defined(__clang_major__) && (__clang_major__ >= 23)
  v8i z8 = {};
  __builtin_amdgcn_tensor_load_to_lds(g0, g1, z4, z4, z8, 0);
#else
  __builtin_amdgcn_tensor_load_to_lds(g0, g1, z4, z4, 0);
#endif
}

// ---------------------------------------------------------------------------
// Problem constants
// ---------------------------------------------------------------------------
#define BB 8
#define NF 64
#define NU 1024
#define DF 64
#define DU 32
#define DH 128
#define HH 8
#define SS 2176  // NF + NU + NF + NU
#define SCALE 0.08838834764831845f  // 1/sqrt(DH)

// ===========================================================================
// Kernel 1: per-head/per-qkv projections -> f16 QKV [3][B][H][S][128].
// The 1/sqrt(DH) softmax scale is folded into Q here.
// ===========================================================================
__global__ __launch_bounds__(256) void qkv_proj_kernel(
    const float* __restrict__ f0, const float* __restrict__ u0,
    const float* __restrict__ f1, const float* __restrict__ u1,
    const float* __restrict__ Wf, const float* __restrict__ bf,
    const float* __restrict__ Wu, const float* __restrict__ bu,
    _Float16* __restrict__ qkv) {
  __shared__ _Float16 Alds[16 * 64];
  __shared__ _Float16 Wlds[64 * 128];

  const int tile = blockIdx.x % 136;            // S/16
  const int h = (blockIdx.x / 136) & 7;
  const int b = (blockIdx.x / (136 * 8)) & 7;
  const int k = blockIdx.x / (136 * 64);        // 0..2 (q,k,v)
  const int s0 = tile * 16;

  const float* x; int din, row0, p;
  if (s0 < NF)                { x = f0 + (size_t)b * NF * DF; din = DF; row0 = s0;             p = 0; }
  else if (s0 < NF + NU)      { x = u0 + (size_t)b * NU * DU; din = DU; row0 = s0 - NF;        p = 0; }
  else if (s0 < 2 * NF + NU)  { x = f1 + (size_t)b * NF * DF; din = DF; row0 = s0 - NF - NU;   p = 1; }
  else                        { x = u1 + (size_t)b * NU * DU; din = DU; row0 = s0 - 2*NF - NU; p = 1; }
  const bool isf = (din == DF);
  const size_t widx = (size_t)((h * 2 + p) * 3 + k);
  const float* W    = (isf ? Wf : Wu) + widx * din * DH;
  const float* bias = (isf ? bf : bu) + widx * DH;

  const int t = threadIdx.x;
  for (int i = t; i < 16 * din; i += 256) {
    int r = i / din, c = i % din;
    Alds[r * 64 + c] = (_Float16)x[(size_t)(row0 + r) * din + c];
  }
  for (int i = t; i < din * DH; i += 256) Wlds[i] = (_Float16)W[i];
  __syncthreads();

  const int w = t >> 5, lane = t & 31;
  v8f acc = {};
  for (int kk = 0; kk < din; kk += 32) {
    v16h a  = load_a_frag(Alds + kk, 64);
    v16h bb = load_b_frag(Wlds + kk * DH + w * 16, DH);
    acc = wmma_f16(a, bb, acc);
  }
  const int col = w * 16 + (lane & 15);
  const float bv = bias[col];
  const float oscale = (k == 0) ? SCALE : 1.0f;   // fold attention scale into Q
  _Float16* outp = qkv + ((((size_t)k * BB + b) * HH + h) * SS + s0) * DH;
#pragma unroll
  for (int r = 0; r < 8; ++r) {
    int rr = r + 8 * (lane >> 4);
    outp[(size_t)rr * DH + col] = (_Float16)((acc[r] + bv) * oscale);
  }
}

// ===========================================================================
// Kernel 2: flash attention per (b,h).  Block = 256 threads (8 waves), each
// wave owns a 16-row q tile (128 q rows / block).  68 key chunks of 32,
// double-buffered in LDS via TDM (tensor_load_to_lds + s_wait_tensorcnt).
//
// Transposed-score mapping: S^T = K·Q^T, so
//   - A operand = K rows, loaded straight from the row-major TDM tile,
//   - B operand = Q^T, staged once per wave,
//   - S^T C-layout (elem r -> key r+8*hh, lane&15 -> q) IS the A-fragment
//     layout required for P·V, so P feeds the second WMMA register-to-register
//     with no LDS transpose; softmax reductions are per-lane + 1 shfl_xor(16).
// ===========================================================================
__global__ __launch_bounds__(256) void attn_kernel(
    const _Float16* __restrict__ qkv, _Float16* __restrict__ attn) {
  __shared__ _Float16 Kbuf[2][32][136];   // row-major keys, 272B pitch
  __shared__ _Float16 Vbuf[2][32][136];
  __shared__ _Float16 Qt[8][128][16];     // per-wave Q^T [d][q]

  const int qblk = blockIdx.x % 17;       // S/128
  const int h = (blockIdx.x / 17) & 7;
  const int b = blockIdx.x / (17 * 8);
  const size_t bh = ((size_t)b * HH + h) * SS * DH;
  const size_t plane = (size_t)BB * HH * SS * DH;
  const _Float16* Qg = qkv + bh;
  const _Float16* Kg = qkv + plane + bh;
  const _Float16* Vg = qkv + 2 * plane + bh;

  const int t = threadIdx.x, w = t >> 5, lane = t & 31;
  const int qbase = qblk * 128 + w * 16;
  const int cidx = lane & 15, hhf = lane >> 4;

  // ---- stage Q^T for this wave (one-time; in-order LDS within the wave) ----
  {
    const int qr = cidx, d0 = hhf * 64;
    const _Float16* qrow = Qg + (size_t)(qbase + qr) * DH + d0;
#pragma unroll
    for (int i = 0; i < 64; i += 8) {
      v8h v = *(const v8h*)(qrow + i);
#pragma unroll
      for (int j = 0; j < 8; ++j) Qt[w][d0 + i + j][qr] = v[j];
    }
  }
  v16h bq[4];
#pragma unroll
  for (int j = 0; j < 4; ++j) bq[j] = load_b_frag(&Qt[w][j * 32][0], 16);

  // ---- TDM prologue: chunk 0 into buffer 0 (wave 0 only, TENSORcnt) ----
  if (w == 0) {
    tdm_load_tile_32x128_f16((unsigned)(size_t)&Kbuf[0][0][0], Kg);
    tdm_load_tile_32x128_f16((unsigned)(size_t)&Vbuf[0][0][0], Vg);
  }

  v8f o[8];
#pragma unroll
  for (int n = 0; n < 8; ++n) { v8f z = {}; o[n] = z; }
  float m_l = -3.0e38f, l_l = 0.f;   // running max / sum for q = lane&15

  for (int cch = 0; cch < 68; ++cch) {
    const int buf = cch & 1;
    if (w == 0) {
      if (cch + 1 < 68) {  // issue next chunk, then wait for current (<=2)
        const size_t goff = (size_t)(cch + 1) * 32 * DH;
        tdm_load_tile_32x128_f16((unsigned)(size_t)&Kbuf[buf ^ 1][0][0], Kg + goff);
        tdm_load_tile_32x128_f16((unsigned)(size_t)&Vbuf[buf ^ 1][0][0], Vg + goff);
        __builtin_amdgcn_s_wait_tensorcnt(2);
      } else {
        __builtin_amdgcn_s_wait_tensorcnt(0);
      }
    }
    __syncthreads();   // chunk cch visible to all waves

    // scores (transposed): S^T[key][q], keys 0-15 -> s0, keys 16-31 -> s1
    v8f s0 = {}, s1 = {};
#pragma unroll
    for (int j = 0; j < 4; ++j) {
      v16h a0 = load_a_frag(&Kbuf[buf][0][j * 32], 136);
      v16h a1 = load_a_frag(&Kbuf[buf][16][j * 32], 136);
      s0 = wmma_f16(a0, bq[j], s0);
      s1 = wmma_f16(a1, bq[j], s1);
    }

    // online softmax: per-lane over 16 keys, + partner lane via shfl_xor(16)
    float mx = fmaxf(s0[0], s1[0]);
#pragma unroll
    for (int r = 1; r < 8; ++r) mx = fmaxf(mx, fmaxf(s0[r], s1[r]));
    mx = fmaxf(mx, __shfl_xor(mx, 16, 32));
    const float mn = fmaxf(m_l, mx);
    const float alpha = __expf(m_l - mn);
    float p0[8], p1[8], rs = 0.f;
#pragma unroll
    for (int r = 0; r < 8; ++r) {
      p0[r] = __expf(s0[r] - mn);
      p1[r] = __expf(s1[r] - mn);
      rs += p0[r] + p1[r];
    }
    rs += __shfl_xor(rs, 16, 32);
    l_l = l_l * alpha + rs;
    m_l = mn;

    // rescale O: alpha lives in lane-space (q=lane&15), O rows in elem-space
#pragma unroll
    for (int r = 0; r < 8; ++r) {
      const float ar = __shfl(alpha, r + 8 * hhf, 16);
#pragma unroll
      for (int n = 0; n < 8; ++n) o[n][r] *= ar;
    }

    // P: S^T C-layout == A-fragment layout for P[q][key] -- pure register pack
    v16h ap;
#pragma unroll
    for (int i = 0; i < 8; ++i) {
      ap[i]     = (_Float16)p0[i];
      ap[8 + i] = (_Float16)p1[i];
    }
#pragma unroll
    for (int n = 0; n < 8; ++n) {
      v16h bv = load_b_frag(&Vbuf[buf][0][n * 16], 136);
      o[n] = wmma_f16(ap, bv, o[n]);
    }
    __syncthreads();   // reads done before next TDM overwrites buf^1
  }

  // epilogue: normalize (l in lane-space -> elem-space), write [B][S][H*DH]
#pragma unroll
  for (int r = 0; r < 8; ++r) {
    const float li = __shfl(l_l, r + 8 * hhf, 16);
    const float inv = 1.0f / li;
    const int s = qbase + r + 8 * hhf;
#pragma unroll
    for (int n = 0; n < 8; ++n) {
      attn[((size_t)b * SS + s) * (HH * DH) + h * DH + n * 16 + cidx] =
          (_Float16)(o[n][r] * inv);
    }
  }
}

// ===========================================================================
// Kernel 3: out = attn[17408,1024] @ Wout[1024,128] + bout  (f32 output)
// ===========================================================================
__global__ __launch_bounds__(256) void outproj_kernel(
    const _Float16* __restrict__ attn, const float* __restrict__ Wout,
    const float* __restrict__ bout, float* __restrict__ out) {
  __shared__ _Float16 Wlds[32 * 128];
  const int t = threadIdx.x, w = t >> 5, lane = t & 31;
  const size_t rbase = (size_t)blockIdx.x * 16;

  v8f acc = {};
  for (int kc = 0; kc < 32; ++kc) {
    const int k0 = kc * 32;
    __syncthreads();
    {  // stage Wout chunk f32 -> f16
      const int r = t >> 3, d0 = (t & 7) * 16;
      const float* src = Wout + (size_t)(k0 + r) * DH + d0;
#pragma unroll
      for (int i = 0; i < 16; ++i) Wlds[r * DH + d0 + i] = (_Float16)src[i];
      if (kc + 1 < 32)  // global_prefetch_b8 of next chunk
        __builtin_prefetch(Wout + (size_t)(k0 + 32 + r) * DH + d0, 0, 0);
    }
    __syncthreads();
    v16h a  = load_a_frag(attn + rbase * (HH * DH) + k0, HH * DH);
    v16h bb = load_b_frag(Wlds + w * 16, DH);
    acc = wmma_f16(a, bb, acc);
  }
  const int col = w * 16 + (lane & 15);
  const float bv = bout[col];
#pragma unroll
  for (int r = 0; r < 8; ++r) {
    int rr = r + 8 * (lane >> 4);
    out[(rbase + rr) * DH + col] = acc[r] + bv;
  }
}

// ===========================================================================
extern "C" void kernel_launch(void* const* d_in, const int* in_sizes, int n_in,
                              void* d_out, int out_size, void* d_ws, size_t ws_size,
                              hipStream_t stream) {
  const float* f0   = (const float*)d_in[0];
  const float* u0   = (const float*)d_in[1];
  const float* f1   = (const float*)d_in[2];
  const float* u1   = (const float*)d_in[3];
  const float* Wf   = (const float*)d_in[4];
  const float* bf   = (const float*)d_in[5];
  const float* Wu   = (const float*)d_in[6];
  const float* bu   = (const float*)d_in[7];
  const float* Wout = (const float*)d_in[8];
  const float* bout = (const float*)d_in[9];
  float* out = (float*)d_out;

  _Float16* qkv  = (_Float16*)d_ws;                       // 3*B*H*S*DH f16
  _Float16* attn = qkv + (size_t)3 * BB * HH * SS * DH;   // B*S*H*DH f16

  qkv_proj_kernel<<<136 * 8 * 8 * 3, 256, 0, stream>>>(f0, u0, f1, u1,
                                                       Wf, bf, Wu, bu, qkv);
  attn_kernel<<<17 * 8 * 8, 256, 0, stream>>>(qkv, attn);
  outproj_kernel<<<1088, 256, 0, stream>>>(attn, Wout, bout, out);
}
